// MultiHeadAttention_1889785610800
// MI455X (gfx1250) — matrix operand order repaired and checked by reading, not compiled, not run on hardware
//
#include <hip/hip_runtime.h>

typedef __attribute__((ext_vector_type(16))) __bf16         v16bf;
typedef __attribute__((ext_vector_type(8)))  float          v8f;
typedef __attribute__((ext_vector_type(8)))  unsigned short u16x8;
typedef __attribute__((ext_vector_type(4)))  float          f32x4;
typedef __attribute__((ext_vector_type(4)))  int            v4i;

union BFrag { u16x8 u[2]; v16bf v; };

__device__ __forceinline__ unsigned short f2bf(float f) {
    unsigned int u = __float_as_uint(f);
    u += 0x7FFFu + ((u >> 16) & 1u);          // round-to-nearest-even
    return (unsigned short)(u >> 16);
}

__device__ __forceinline__ v8f wmma_bf16(v16bf a, v16bf b, v8f c) {
    return __builtin_amdgcn_wmma_f32_16x16x32_bf16(false, a, false, b, (short)0, c, false, false);
}

// ---- CDNA5 async global->LDS copy (16B), ASYNCcnt-tracked --------------------
__device__ __forceinline__ void async_cp16(const unsigned short* g, unsigned short* l) {
#if __has_builtin(__builtin_amdgcn_global_load_async_to_lds_b128)
    __builtin_amdgcn_global_load_async_to_lds_b128(
        (__attribute__((address_space(1))) v4i*)g,
        (__attribute__((address_space(3))) v4i*)l, 0, 0);
#else
    *(u16x8*)l = *(const u16x8*)g;
#endif
}

__device__ __forceinline__ void async_wait0() {
#if __has_builtin(__builtin_amdgcn_s_wait_asynccnt)
    __builtin_amdgcn_s_wait_asynccnt(0);
#else
    asm volatile("s_wait_asynccnt 0x0" ::: "memory");
#endif
}

// ---------------------------------------------------------------------------
// fp32 -> bf16 bulk convert (8 elements / thread)
// ---------------------------------------------------------------------------
__global__ __launch_bounds__(256) void convert_bf16_kernel(
    const float* __restrict__ src, unsigned short* __restrict__ dst) {
    const size_t i = ((size_t)blockIdx.x * 256 + threadIdx.x) * 8;
    f32x4 x0 = *(const f32x4*)(src + i);
    f32x4 x1 = *(const f32x4*)(src + i + 4);
    u16x8 o;
#pragma unroll
    for (int j = 0; j < 4; ++j) { o[j] = f2bf(x0[j]); o[4 + j] = f2bf(x1[j]); }
    *(u16x8*)(dst + i) = o;
}

// ---------------------------------------------------------------------------
// mask (int 0/1) -> additive bias (0 / -1e9)
// ---------------------------------------------------------------------------
__global__ void maskbias_kernel(const int* __restrict__ mask,
                                float* __restrict__ mb, int n) {
    int i = blockIdx.x * blockDim.x + threadIdx.x;
    if (i < n) mb[i] = mask[i] ? 0.0f : -1.0e9f;
}

// ---------------------------------------------------------------------------
// GEMM: Y[m,n] = sum_k A[m,k]*W[n,k] + bias[n]   (A,W bf16, fp32 accumulate)
//   MODE 0: out bf16 [B,H,S,DK]   MODE 1: out bf16 [B,H,DK,S]   MODE 2: fp32 [M,N]
// 128x128 block tile, K-step 32, 8 waves x (64x32), double-buffered async LDS.
// ---------------------------------------------------------------------------
template<int MODE>
__global__ __launch_bounds__(256) void gemm_bf16_kernel(
    const unsigned short* __restrict__ Ab, const unsigned short* __restrict__ Wb,
    const float* __restrict__ bias, void* __restrict__ Out)
{
    constexpr int K   = 1024;
    constexpr int LDT = 40;                    // LDS row stride (halves), 16B-aligned
    __shared__ unsigned short As[2][128 * LDT];
    __shared__ unsigned short Ws[2][128 * LDT];

    const int t    = threadIdx.x;
    const int lane = t & 31;
    const int wid  = t >> 5;
    const int l15  = lane & 15;
    const int lh   = lane >> 4;
    const int m0   = blockIdx.x * 128;
    const int n0   = blockIdx.y * 128;
    const int wm   = (wid & 1) * 64;
    const int wn   = (wid >> 1) * 32;

    auto stage = [&](int buf, int k0) {
#pragma unroll
        for (int i = 0; i < 2; ++i) {
            const int seg = t * 2 + i;                 // 0..511
            const int r = seg >> 2, c = (seg & 3) * 8; // row, half-offset
            async_cp16(Ab + (size_t)(m0 + r) * K + k0 + c, &As[buf][r * LDT + c]);
            async_cp16(Wb + (size_t)(n0 + r) * K + k0 + c, &Ws[buf][r * LDT + c]);
        }
    };

    v8f acc[4][2];
    const v8f vzero = {0.f,0.f,0.f,0.f,0.f,0.f,0.f,0.f};
#pragma unroll
    for (int i = 0; i < 4; ++i)
#pragma unroll
        for (int j = 0; j < 2; ++j) acc[i][j] = vzero;

    stage(0, 0);
    int cur = 0;
    for (int k0 = 0; k0 < K; k0 += 32) {
        async_wait0();
        __syncthreads();                       // buf[cur] visible to all waves
        if (k0 + 32 < K) stage(cur ^ 1, k0 + 32);

        BFrag af[4], bf[2];
#pragma unroll
        for (int mt = 0; mt < 4; ++mt) {
            const unsigned short* base = &As[cur][(wm + mt * 16 + l15) * LDT];
            af[mt].u[0] = *(const u16x8*)(base + lh * 8);
            af[mt].u[1] = *(const u16x8*)(base + 16 + lh * 8);
        }
#pragma unroll
        for (int nt = 0; nt < 2; ++nt) {
            const unsigned short* base = &Ws[cur][(wn + nt * 16 + l15) * LDT] + lh * 8;
            bf[nt].u[0] = *(const u16x8*)(base);
            bf[nt].u[1] = *(const u16x8*)(base + 16);
        }
#pragma unroll
        for (int mt = 0; mt < 4; ++mt)
#pragma unroll
            for (int nt = 0; nt < 2; ++nt)
                acc[mt][nt] = wmma_bf16(af[mt].v, bf[nt].v, acc[mt][nt]);
        __syncthreads();                       // all waves done reading buf[cur]
        cur ^= 1;
    }

#pragma unroll
    for (int nt = 0; nt < 2; ++nt) {
        const int n = n0 + wn + nt * 16 + l15;
        const float bv = bias[n];
#pragma unroll
        for (int mt = 0; mt < 4; ++mt)
#pragma unroll
            for (int v = 0; v < 8; ++v) {
                const int m = m0 + wm + mt * 16 + lh * 8 + v;
                const float y = acc[mt][nt][v] + bv;
                if constexpr (MODE == 2) {
                    ((float*)Out)[(size_t)m * 1024 + n] = y;
                } else {
                    const int b = m >> 11, s = m & 2047;   // S=2048
                    const int h = n >> 6,  dk = n & 63;    // DK=64
                    if constexpr (MODE == 0)
                        ((unsigned short*)Out)[((((size_t)b * 16 + h) * 2048 + s) << 6) + dk] = f2bf(y);
                    else
                        ((unsigned short*)Out)[((((size_t)b * 16 + h) * 64 + dk) << 11) + s] = f2bf(y);
                }
            }
    }
}

// ---------------------------------------------------------------------------
// Flash attention. Q/K [B,H,S,DK] bf16, V^T [B,H,DK,S] bf16, out bf16 [B,S,D].
// Block = 8 waves, 128 q rows; K/V tiles (128 keys) staged async into LDS,
// double-buffered, shared by all waves. Each wave owns 16 q rows.
// Dynamic LDS layout (halves): Kt[2][128*72] | Vt[2][64*136] | P[8][16*136]
// ---------------------------------------------------------------------------
__global__ __launch_bounds__(256) void attn_kernel(
    const unsigned short* __restrict__ Qh,
    const unsigned short* __restrict__ Kh,
    const unsigned short* __restrict__ VhT,
    const float* __restrict__ mb,
    unsigned short* __restrict__ Out)
{
    constexpr int S = 2048, DK = 64, Dm = 1024;
    constexpr int KT = 72, VT = 136, PLD = 136;      // LDS row strides (halves)
    extern __shared__ unsigned short smem[];
    unsigned short* Kt = smem;                        // 2 * 9216
    unsigned short* Vt = smem + 2 * 128 * KT;         // 2 * 8704
    unsigned short* Pb = Vt + 2 * 64 * VT;            // 8 * 16 * 136

    const int t    = threadIdx.x;
    const int lane = t & 31;
    const int wid  = t >> 5;
    const int l15  = lane & 15;
    const int lh   = lane >> 4;
    const int bh   = blockIdx.x >> 4;
    const int qblk = blockIdx.x & 15;
    const int b    = bh >> 4;
    const int h    = bh & 15;
    const int q0   = qblk * 128 + wid * 16;
    unsigned short* Pw = Pb + wid * 16 * PLD;

    const float*          mbase = mb  + (size_t)b  * S;
    const unsigned short* Kbase = Kh  + (size_t)bh * S  * DK;
    const unsigned short* Vbase = VhT + (size_t)bh * DK * S;

    auto stageKV = [&](int buf, int kb) {
#pragma unroll
        for (int i = 0; i < 4; ++i) {                 // K tile: 128 x 64 halves
            const int seg = t * 4 + i;                // 0..1023
            const int r = seg >> 3, c = (seg & 7) * 8;
            async_cp16(Kbase + (size_t)(kb + r) * DK + c, Kt + buf * 128 * KT + r * KT + c);
        }
#pragma unroll
        for (int i = 0; i < 4; ++i) {                 // V tile: 64 x 128 halves
            const int seg = t * 4 + i;
            const int d = seg >> 4, c = (seg & 15) * 8;
            async_cp16(Vbase + (size_t)d * S + kb + c, Vt + buf * 64 * VT + d * VT + c);
        }
    };

    // resident Q fragments (A layout): 16 rows x 64 k
    BFrag qf[2];
    const unsigned short* Qbase = Qh + ((size_t)bh * S + q0 + l15) * DK;
#pragma unroll
    for (int f = 0; f < 2; ++f) {
        qf[f].u[0] = *(const u16x8*)(Qbase + f * 32 + lh * 8);
        qf[f].u[1] = *(const u16x8*)(Qbase + f * 32 + 16 + lh * 8);
    }

    const v8f vzero = {0.f,0.f,0.f,0.f,0.f,0.f,0.f,0.f};
    v8f Oacc[4];
#pragma unroll
    for (int nt = 0; nt < 4; ++nt) Oacc[nt] = vzero;
    float rm[8], rs[8];
#pragma unroll
    for (int v = 0; v < 8; ++v) { rm[v] = -1.0e30f; rs[v] = 0.0f; }

    stageKV(0, 0);
    int cur = 0;
    for (int kb = 0; kb < S; kb += 128) {
        async_wait0();
        __syncthreads();                              // K/V tile ready
        if (kb + 128 < S) stageKV(cur ^ 1, kb + 128);
        const unsigned short* Ktc = Kt + cur * 128 * KT;
        const unsigned short* Vtc = Vt + cur * 64 * VT;

        // ---- scores S = Q K^T : 16 WMMA (K frags from LDS) ----
        v8f Sacc[8];
#pragma unroll
        for (int j = 0; j < 8; ++j) Sacc[j] = vzero;
#pragma unroll
        for (int j = 0; j < 8; ++j) {
            const unsigned short* kp = Ktc + (j * 16 + l15) * KT;
            BFrag k0f, k1f;
            k0f.u[0] = *(const u16x8*)(kp + lh * 8);
            k0f.u[1] = *(const u16x8*)(kp + lh * 8 + 16);
            k1f.u[0] = *(const u16x8*)(kp + 32 + lh * 8);
            k1f.u[1] = *(const u16x8*)(kp + 32 + lh * 8 + 16);
            Sacc[j] = wmma_bf16(qf[0].v, k0f.v, Sacc[j]);
            Sacc[j] = wmma_bf16(qf[1].v, k1f.v, Sacc[j]);
        }

        // ---- scale + mask ----
        float mv[8];
#pragma unroll
        for (int j = 0; j < 8; ++j) mv[j] = mbase[kb + j * 16 + l15];
#pragma unroll
        for (int j = 0; j < 8; ++j)
#pragma unroll
            for (int v = 0; v < 8; ++v)
                Sacc[j][v] = Sacc[j][v] * 0.125f + mv[j];

        // ---- online softmax (row reductions across 16 column-lanes) ----
        float pm[8];
#pragma unroll
        for (int v = 0; v < 8; ++v) {
            pm[v] = Sacc[0][v];
#pragma unroll
            for (int j = 1; j < 8; ++j) pm[v] = fmaxf(pm[v], Sacc[j][v]);
        }
#pragma unroll
        for (int off = 1; off < 16; off <<= 1)
#pragma unroll
            for (int v = 0; v < 8; ++v)
                pm[v] = fmaxf(pm[v], __shfl_xor(pm[v], off, 32));

        float cf[8];
#pragma unroll
        for (int v = 0; v < 8; ++v) {
            const float nm = fmaxf(rm[v], pm[v]);
            cf[v] = __expf(rm[v] - nm);
            rm[v] = nm;
        }

        float ps[8];
#pragma unroll
        for (int v = 0; v < 8; ++v) ps[v] = 0.0f;
#pragma unroll
        for (int j = 0; j < 8; ++j)
#pragma unroll
            for (int v = 0; v < 8; ++v) {
                const float p = __expf(Sacc[j][v] - rm[v]);
                ps[v] += p;
                Pw[(lh * 8 + v) * PLD + j * 16 + l15] = f2bf(p);
            }
#pragma unroll
        for (int off = 1; off < 16; off <<= 1)
#pragma unroll
            for (int v = 0; v < 8; ++v)
                ps[v] += __shfl_xor(ps[v], off, 32);
#pragma unroll
        for (int v = 0; v < 8; ++v) rs[v] = rs[v] * cf[v] + ps[v];
#pragma unroll
        for (int nt = 0; nt < 4; ++nt)
#pragma unroll
            for (int v = 0; v < 8; ++v) Oacc[nt][v] *= cf[v];

        // ---- O += P V : 16 WMMA (P and V frags from LDS) ----
#pragma unroll
        for (int kk = 0; kk < 4; ++kk) {
            BFrag pf;
            const unsigned short* pb = Pw + l15 * PLD + kk * 32;
            pf.u[0] = *(const u16x8*)(pb + lh * 8);
            pf.u[1] = *(const u16x8*)(pb + 16 + lh * 8);
#pragma unroll
            for (int nt = 0; nt < 4; ++nt) {
                BFrag vf;
                const unsigned short* vp = Vtc + (nt * 16 + l15) * VT + kk * 32 + lh * 8;
                vf.u[0] = *(const u16x8*)(vp);
                vf.u[1] = *(const u16x8*)(vp + 16);
                Oacc[nt] = wmma_bf16(pf.v, vf.v, Oacc[nt]);
            }
        }
        __syncthreads();                              // done with buf[cur]
        cur ^= 1;
    }

    float inv[8];
#pragma unroll
    for (int v = 0; v < 8; ++v) inv[v] = 1.0f / rs[v];
#pragma unroll
    for (int nt = 0; nt < 4; ++nt)
#pragma unroll
        for (int v = 0; v < 8; ++v) {
            const int row = lh * 8 + v;
            const size_t idx = ((size_t)b * S + q0 + row) * Dm + h * DK + nt * 16 + l15;
            Out[idx] = f2bf(Oacc[nt][v] * inv[v]);
        }
}

// ---------------------------------------------------------------------------
extern "C" void kernel_launch(void* const* d_in, const int* in_sizes, int n_in,
                              void* d_out, int out_size, void* d_ws, size_t ws_size,
                              hipStream_t stream) {
    (void)in_sizes; (void)n_in; (void)out_size; (void)ws_size;
    // inputs: 0 query 1 key 2 value 3 mask 4 wq 5 bq 6 wk 7 bk 8 wv 9 bv 10 wo 11 bo
    char* ws = (char*)d_ws;
    const size_t MB = (size_t)1 << 20;
    unsigned short* qb  = (unsigned short*)(ws);             // 16 MB bf16 activations
    unsigned short* kbuf= (unsigned short*)(ws + 16 * MB);
    unsigned short* vb  = (unsigned short*)(ws + 32 * MB);
    unsigned short* wqb = (unsigned short*)(ws + 48 * MB);   // 2 MB bf16 weights
    unsigned short* wkb = (unsigned short*)(ws + 50 * MB);
    unsigned short* wvb = (unsigned short*)(ws + 52 * MB);
    unsigned short* wob = (unsigned short*)(ws + 54 * MB);
    unsigned short* Qh  = (unsigned short*)(ws + 56 * MB);   // [B,H,S,DK]
    unsigned short* Kh  = (unsigned short*)(ws + 72 * MB);   // [B,H,S,DK]
    unsigned short* VhT = (unsigned short*)(ws + 88 * MB);   // [B,H,DK,S]
    unsigned short* AO  = (unsigned short*)(ws + 104 * MB);  // [B,S,D]
    float*          mbF = (float*)         (ws + 120 * MB);

    const int NA = 4 * 2048 * 1024;   // activation elements
    const int NW = 1024 * 1024;       // weight elements
    convert_bf16_kernel<<<NA / 2048, 256, 0, stream>>>((const float*)d_in[0], qb);
    convert_bf16_kernel<<<NA / 2048, 256, 0, stream>>>((const float*)d_in[1], kbuf);
    convert_bf16_kernel<<<NA / 2048, 256, 0, stream>>>((const float*)d_in[2], vb);
    convert_bf16_kernel<<<NW / 2048, 256, 0, stream>>>((const float*)d_in[4], wqb);
    convert_bf16_kernel<<<NW / 2048, 256, 0, stream>>>((const float*)d_in[6], wkb);
    convert_bf16_kernel<<<NW / 2048, 256, 0, stream>>>((const float*)d_in[8], wvb);
    convert_bf16_kernel<<<NW / 2048, 256, 0, stream>>>((const float*)d_in[10], wob);

    maskbias_kernel<<<(4 * 2048 + 255) / 256, 256, 0, stream>>>(
        (const int*)d_in[3], mbF, 4 * 2048);

    dim3 gg(64, 8), bb(256);
    gemm_bf16_kernel<0><<<gg, bb, 0, stream>>>(qb,   wqb, (const float*)d_in[5], Qh);
    gemm_bf16_kernel<0><<<gg, bb, 0, stream>>>(kbuf, wkb, (const float*)d_in[7], Kh);
    gemm_bf16_kernel<1><<<gg, bb, 0, stream>>>(vb,   wvb, (const float*)d_in[9], VhT);

    const size_t attn_lds = (2 * 128 * 72 + 2 * 64 * 136 + 8 * 16 * 136) * sizeof(unsigned short);
    attn_kernel<<<1024, 256, attn_lds, stream>>>(Qh, Kh, VhT, mbF, AO);

    gemm_bf16_kernel<2><<<gg, bb, 0, stream>>>(AO, wob, (const float*)d_in[11], d_out);
}
